// GraphNetV1_893353198489
// MI455X (gfx1250) — matrix operand order repaired and checked by reading, not compiled, not run on hardware
//
#include <hip/hip_runtime.h>
#include <hip/hip_bf16.h>

typedef __attribute__((ext_vector_type(16))) _Float16 v16h;
typedef __attribute__((ext_vector_type(8)))  float    v8f;

#define N_NODES 50000
#define N_EDGES 800000
#define D_FEAT  100
#define K_PAD   128   // 4 WMMA k-steps of 32
#define N_PAD   112   // 7 WMMA n-tiles of 16

// ---- workspace layout (bytes, 256-aligned) ----
static constexpr size_t OFF_NORM = 0;                                   // 1 f32
static constexpr size_t OFF_DEG  = 256;                                 // 50000 f32
static constexpr size_t OFF_DINV = 200448;                              // 50000 f32
static constexpr size_t OFF_AGGA = 400640;                              // 50000*112 f32
static constexpr size_t OFF_AGGB = OFF_AGGA + (size_t)N_NODES*N_PAD*4;  // 50000*112 f32
static constexpr size_t OFF_H    = OFF_AGGB + (size_t)N_NODES*N_PAD*4;  // 50000*112 f32
static constexpr size_t OFF_X16  = OFF_H    + (size_t)N_NODES*N_PAD*4;  // 50000*128 f16 (frag order)
static constexpr size_t OFF_BPAD = OFF_X16  + (size_t)N_NODES*K_PAD*2;  // 128*112 f16 (frag order)
static constexpr size_t ZERO_FLOATS = OFF_H / 4;  // zero norm,deg,dinv,aggA,aggB

// ---------------- utility kernels ----------------
__global__ void zero_kernel(float* p, size_t n) {
  size_t i = (size_t)blockIdx.x * blockDim.x + threadIdx.x;
  size_t stride = (size_t)gridDim.x * blockDim.x;
  for (; i < n; i += stride) p[i] = 0.0f;
}

__global__ void sumsq_kernel(const float* __restrict__ x, float* __restrict__ accum, size_t n) {
  size_t i = (size_t)blockIdx.x * blockDim.x + threadIdx.x;
  size_t stride = (size_t)gridDim.x * blockDim.x;
  float s = 0.0f;
  for (; i < n; i += stride) { float v = x[i]; s += v * v; }
  #pragma unroll
  for (int m = 16; m >= 1; m >>= 1) s += __shfl_xor(s, m, 32);
  if ((threadIdx.x & 31) == 0) unsafeAtomicAdd(accum, s);
}

__global__ void degree_kernel(const int* __restrict__ dst, float* __restrict__ deg, int E) {
  int e = blockIdx.x * blockDim.x + threadIdx.x;
  if (e < E) unsafeAtomicAdd(&deg[dst[e]], 1.0f);
}

__global__ void dinv_kernel(const float* __restrict__ deg, float* __restrict__ dinv, int N) {
  int i = blockIdx.x * blockDim.x + threadIdx.x;
  if (i >= N) return;
  float d = deg[i] + 1.0f;               // self-loop included
  dinv[i] = (d > 0.0f) ? rsqrtf(d) : 0.0f;
}

// ---------------- fragment-order packing ----------------
// A fragment layout (CDNA5 ISA 7.12.2, 16-bit A 16x32):
//   chunk c = mt*4 + kt holds 32 lanes x 16 halves (512 f16, 1 KB)
//   lane L: row = mt*16 + (L&15)
//   half h: v = h>>1; k = kt*32 + 2*v + (v<4?0:8) + (L>=16?8:0) + (h&1)
__global__ void convert_pad_frag_kernel(const float* __restrict__ X, int srcStride,
                                        const float* __restrict__ sumsqPtr,
                                        _Float16* __restrict__ O, int N) {
  int idx = blockIdx.x * blockDim.x + threadIdx.x;
  if (idx >= N * K_PAD) return;
  int c    = idx >> 9;           // (mt*4 + kt)
  int lane = (idx >> 4) & 31;
  int h    = idx & 15;
  int mt = c >> 2, kt = c & 3;
  int hv = h >> 1, hp = h & 1;
  int row = mt * 16 + (lane & 15);
  int k   = kt * 32 + 2 * hv + (hv < 4 ? 0 : 8) + (lane >> 4) * 8 + hp;
  float scale = sumsqPtr ? rsqrtf(*sumsqPtr) : 1.0f;
  float v = (k < D_FEAT) ? X[(size_t)row * srcStride + k] * scale : 0.0f;
  O[idx] = (_Float16)v;
}

// B fragment layout (16-bit B 32x16):
//   chunk c = nt*4 + kt; lane L: n = nt*16 + (L&15)
//   half h: k = kt*32 + (L>=16?8:0) + (h<8 ? h : h+8)
__global__ void packW_frag_kernel(const float* __restrict__ W, _Float16* __restrict__ B) {
  int idx = blockIdx.x * blockDim.x + threadIdx.x;
  if (idx >= 7 * 4 * 512) return;
  int c    = idx >> 9;
  int lane = (idx >> 4) & 31;
  int h    = idx & 15;
  int nt = c >> 2, kt = c & 3;
  int n = nt * 16 + (lane & 15);
  int k = kt * 32 + (lane >> 4) * 8 + (h < 8 ? h : h + 8);
  float v = (k < D_FEAT && n < D_FEAT) ? W[k * D_FEAT + n] : 0.0f;
  B[idx] = (_Float16)v;
}

// ---------------- WMMA GEMM: C[M,112] = A[M,128] * B[128,112] ----------------
// 224 threads = 7 waves; wave = n-tile, blockIdx.x = m-tile. Operands are
// pre-packed in fragment order: each lane loads one contiguous 32 B v16h.
__global__ void __launch_bounds__(224) gemm_wmma_kernel(const _Float16* __restrict__ A,
                                                        const _Float16* __restrict__ B,
                                                        float* __restrict__ C) {
  int wave   = threadIdx.x >> 5;
  int lane   = threadIdx.x & 31;
  int laneHi = (lane >> 4) & 1;
  int lm     = lane & 15;
  int m0     = blockIdx.x << 4;
  int n0     = wave << 4;

  const v16h* Av = (const v16h*)A + ((size_t)blockIdx.x * 4) * 32 + lane;
  const v16h* Bv = (const v16h*)B + ((size_t)wave * 4) * 32 + lane;

  v8f acc = {};
  #pragma unroll
  for (int kt = 0; kt < 4; ++kt) {
    v16h a = Av[kt * 32];
    v16h b = Bv[kt * 32];
    acc = __builtin_amdgcn_wmma_f32_16x16x32_f16(false, a, false, b,
                                                 (short)0, acc, false, false);
  }
  // C/D layout: VGPR v -> M = v + laneHi*8, N = lane%16
  #pragma unroll
  for (int v = 0; v < 8; ++v)
    C[(size_t)(m0 + v + laneHi * 8) * N_PAD + n0 + lm] = acc[v];
}

// ---------------- message passing ----------------
// one wave per edge: agg[dst] += dinv[src]*dinv[dst] * h[src]
__global__ void edge_agg_kernel(const int* __restrict__ src, const int* __restrict__ dst,
                                const float* __restrict__ dinv, const float* __restrict__ h,
                                float* __restrict__ agg, int E) {
  int gw = (int)(((size_t)blockIdx.x * blockDim.x + threadIdx.x) >> 5);
  int lane = threadIdx.x & 31;
  if (gw >= E) return;
  int s = src[gw], d = dst[gw];
  float w = dinv[s] * dinv[d];
  const float* hs = h + (size_t)s * N_PAD;
  __builtin_prefetch(hs, 0, 0);                       // global_prefetch_b8
  float* ad = agg + (size_t)d * N_PAD;
  for (int k = lane; k < D_FEAT; k += 32)
    unsafeAtomicAdd(&ad[k], w * hs[k]);
}

// self-loop contribution (dinv[i]^2 == 1/deg) + bias, done after edge atomics
__global__ void selfloop_bias_kernel(const float* __restrict__ h, const float* __restrict__ dinv,
                                     const float* __restrict__ bias, float* __restrict__ agg,
                                     int N) {
  int i = blockIdx.x;
  int k = threadIdx.x;
  if (i >= N || k >= D_FEAT) return;
  float di = dinv[i];
  agg[(size_t)i * N_PAD + k] += h[(size_t)i * N_PAD + k] * di * di + bias[k];
}

// ---------------- MLP head: relu(X@W0+b0)@W1+b1 -> out[N] ----------------
__global__ void mlp_kernel(const float* __restrict__ X, const float* __restrict__ W0,
                           const float* __restrict__ b0, const float* __restrict__ W1,
                           const float* __restrict__ b1, float* __restrict__ out, int N) {
  int i = blockIdx.x * blockDim.x + threadIdx.x;
  if (i >= N) return;
  float acc[10];
  #pragma unroll
  for (int j = 0; j < 10; ++j) acc[j] = b0[j];
  const float* xr = X + (size_t)i * N_PAD;
  for (int k = 0; k < D_FEAT; ++k) {
    float xv = xr[k];
    #pragma unroll
    for (int j = 0; j < 10; ++j) acc[j] += xv * W0[k * 10 + j];
  }
  float o = b1[0];
  #pragma unroll
  for (int j = 0; j < 10; ++j) o += fmaxf(acc[j], 0.0f) * W1[j];
  out[i] = o;
}

// ---------------- launcher ----------------
extern "C" void kernel_launch(void* const* d_in, const int* in_sizes, int n_in,
                              void* d_out, int out_size, void* d_ws, size_t ws_size,
                              hipStream_t stream) {
  const float* x      = (const float*)d_in[0];
  const int*   eidx   = (const int*)d_in[1];
  const float* convW0 = (const float*)d_in[2];
  const float* convb0 = (const float*)d_in[3];
  const float* convW1 = (const float*)d_in[4];
  const float* convb1 = (const float*)d_in[5];
  const float* linW0  = (const float*)d_in[6];
  const float* linb0  = (const float*)d_in[7];
  const float* linW1  = (const float*)d_in[8];
  const float* linb1  = (const float*)d_in[9];
  float* out = (float*)d_out;

  const int N = in_sizes[0] / D_FEAT;   // 50000
  const int E = in_sizes[1] / 2;        // 800000
  const int* esrc = eidx;
  const int* edst = eidx + E;

  char* ws = (char*)d_ws;
  float*    normAcc = (float*)(ws + OFF_NORM);
  float*    deg     = (float*)(ws + OFF_DEG);
  float*    dinv    = (float*)(ws + OFF_DINV);
  float*    aggA    = (float*)(ws + OFF_AGGA);
  float*    aggB    = (float*)(ws + OFF_AGGB);
  float*    hbuf    = (float*)(ws + OFF_H);
  _Float16* x16     = (_Float16*)(ws + OFF_X16);
  _Float16* Bpad    = (_Float16*)(ws + OFF_BPAD);

  // 1. zero accumulators (norm, deg, dinv, aggA, aggB)
  zero_kernel<<<2048, 256, 0, stream>>>((float*)ws, ZERO_FLOATS);
  // 2. ||x||^2
  sumsq_kernel<<<1024, 256, 0, stream>>>(x, normAcc, (size_t)N * D_FEAT);
  // 3-4. degree, d^{-1/2}
  degree_kernel<<<(E + 255) / 256, 256, 0, stream>>>(edst, deg, E);
  dinv_kernel<<<(N + 255) / 256, 256, 0, stream>>>(deg, dinv, N);

  // ---- conv0 ----
  convert_pad_frag_kernel<<<(N * K_PAD + 255) / 256, 256, 0, stream>>>(x, D_FEAT, normAcc, x16, N);
  packW_frag_kernel<<<(7 * 4 * 512 + 255) / 256, 256, 0, stream>>>(convW0, Bpad);
  gemm_wmma_kernel<<<N / 16, 224, 0, stream>>>(x16, Bpad, hbuf);
  edge_agg_kernel<<<(int)(((size_t)E * 32 + 255) / 256), 256, 0, stream>>>(esrc, edst, dinv, hbuf, aggA, E);
  selfloop_bias_kernel<<<N, 128, 0, stream>>>(hbuf, dinv, convb0, aggA, N);

  // ---- conv1 ----
  convert_pad_frag_kernel<<<(N * K_PAD + 255) / 256, 256, 0, stream>>>(aggA, N_PAD, nullptr, x16, N);
  packW_frag_kernel<<<(7 * 4 * 512 + 255) / 256, 256, 0, stream>>>(convW1, Bpad);
  gemm_wmma_kernel<<<N / 16, 224, 0, stream>>>(x16, Bpad, hbuf);
  edge_agg_kernel<<<(int)(((size_t)E * 32 + 255) / 256), 256, 0, stream>>>(esrc, edst, dinv, hbuf, aggB, E);
  selfloop_bias_kernel<<<N, 128, 0, stream>>>(hbuf, dinv, convb1, aggB, N);

  // ---- MLP head ----
  mlp_kernel<<<(N + 255) / 256, 256, 0, stream>>>(aggB, linW0, linb0, linW1, linb1, out, N);
}
